// ViT_2783138808322
// MI455X (gfx1250) — matrix operand order repaired and checked by reading, not compiled, not run on hardware
//
#include <hip/hip_runtime.h>
#include <hip/hip_bf16.h>

// ---------------- model constants ----------------
#define BATCHN 32
#define SEQ    197
#define SEQP   208                 // SEQ padded to multiple of 16 (vector-load friendly)
#define HIDD   768
#define LYR    12
#define NHD    12
#define DHD    64
#define INTERD 3072
#define NCLS   1000
#define NPP    196                 // patches per image
#define TOK    (BATCHN * SEQ)      // 6304
#define NPATCH (BATCHN * NPP)      // 6272

typedef __attribute__((ext_vector_type(16))) __bf16 v16bf;
typedef __attribute__((ext_vector_type(8)))  __bf16 v8bf;
typedef __attribute__((ext_vector_type(8)))  float  v8f;

__device__ __forceinline__ int imin(int a, int b) { return a < b ? a : b; }

// ---------------- WMMA GEMM ----------------
// C[M,N] = epilogue( alpha * A[M,K] @ B[K,N] )
// A,B bf16 row-major, C f32 and/or bf16.
// REQUIREMENTS (all call sites satisfy them): lda%8==0, ldb%8==0, N%8==0, K%8==0.
//   - A rows are clamped to M-1 (rows >= M never reach C: epilogue masks them)
//   - B cols are clamped to N-8 chunk (cols >= N never reach C)
//   - K-tail chunks are zeroed in A only => B's clamped K-tail rows multiply by 0
// This keeps ALL global staging as aligned v8bf loads with zero divergent branches.
// Batched via z: off = (z/bdiv)*S1 + (z%bdiv)*S2 per operand.
// 128 threads = 4 waves; block tile 64x64; each wave computes a 32x32 C tile as
// 2x2 16x16 WMMA sub-tiles. K-loop is software-pipelined: next tile's global
// loads are issued before the current tile's fragment loads + 4x WMMA.
struct GemmP {
  const __bf16* A; const __bf16* B; const float* bias; const float* resid;
  float* Cf; __bf16* Cb;
  int M, N, K, lda, ldb, ldc;
  int bdiv;
  long aS1, aS2, bS1, bS2, cS1, cS2;
  float alpha;
};

template<bool BIAS, bool GELU, bool RESID, bool SF32, bool SBF16>
__global__ __launch_bounds__(128) void gemm_wmma_k(GemmP p) {
  // 40-half (80B) row stride -> all fragment reads are 16B-aligned ds_load_b128
  __shared__ __bf16 As[64][40];   // [m][k]
  __shared__ __bf16 Bs[64][40];   // transposed: [n][k]
  const int z = blockIdx.z;
  const long aOff = (long)(z / p.bdiv) * p.aS1 + (long)(z % p.bdiv) * p.aS2;
  const long bOff = (long)(z / p.bdiv) * p.bS1 + (long)(z % p.bdiv) * p.bS2;
  const long cOff = (long)(z / p.bdiv) * p.cS1 + (long)(z % p.bdiv) * p.cS2;
  const __bf16* Ag = p.A + aOff;
  const __bf16* Bg = p.B + bOff;
  const int bm = blockIdx.y * 64;
  const int bn = blockIdx.x * 64;
  const int tid  = threadIdx.x;
  const int lane = tid & 31;
  const int w    = tid >> 5;               // 0..3
  const int wm   = w >> 1;                 // 0..1 : 32-row strip
  const int wn   = w & 1;                  // 0..1 : 32-col strip
  const int fm   = lane & 15;
  const int akb  = (lane < 16) ? 0 : 16;   // A-frag K base (contiguous k, matches B)
  const int bkb  = (lane < 16) ? 0 : 16;   // B-frag K base (contiguous k, matches A)
  // staging coords (128 threads)
  const int ar_ld = tid >> 1, ac_ld = (tid & 1) * 16;  // A: 64 rows x 32 k
  const int br_ld = tid >> 2, bc_ld = (tid & 3) * 16;  // B: 32 k  x 64 n
  // clamped (always in-bounds, always 16B-aligned) global addresses
  const long arow_base = (long)imin(bm + ar_ld, p.M - 1) * p.lda;
  const int  gn0 = imin(bn + bc_ld,     p.N - 8);
  const int  gn1 = imin(bn + bc_ld + 8, p.N - 8);

  v8bf vz;
#pragma unroll
  for (int i = 0; i < 8; ++i) vz[i] = (__bf16)0.0f;

  v8f acc00 = {}, acc01 = {}, acc10 = {}, acc11 = {};
  v8bf ra0, ra1, rb0, rb1;

  // issue global loads for a K-step (branchless: clamp + vector select)
  auto gstage = [&](int k0) {
    const int gc0 = k0 + ac_ld, gc1 = k0 + ac_ld + 8;
    ra0 = *(const v8bf*)&Ag[arow_base + imin(gc0, p.K - 8)];
    ra1 = *(const v8bf*)&Ag[arow_base + imin(gc1, p.K - 8)];
    if (gc0 >= p.K) ra0 = vz;              // zero A's K-tail (B tail rides on 0*x)
    if (gc1 >= p.K) ra1 = vz;
    const long brow = (long)imin(k0 + br_ld, p.K - 1) * p.ldb;
    rb0 = *(const v8bf*)&Bg[brow + gn0];
    rb1 = *(const v8bf*)&Bg[brow + gn1];
  };

  gstage(0);
  for (int k0 = 0; k0 < p.K; k0 += 32) {
    // ---- commit staged registers to LDS ----
    *(v8bf*)&As[ar_ld][ac_ld]     = ra0;
    *(v8bf*)&As[ar_ld][ac_ld + 8] = ra1;
#pragma unroll
    for (int i = 0; i < 8; ++i) Bs[bc_ld + i][br_ld]     = rb0[i];
#pragma unroll
    for (int i = 0; i < 8; ++i) Bs[bc_ld + 8 + i][br_ld] = rb1[i];
    __syncthreads();
    // ---- issue next tile's global loads (latency hidden behind WMMAs) ----
    if (k0 + 32 < p.K) gstage(k0 + 32);
    // ---- fragments (b128 LDS loads) + 4x WMMA ----
    v16bf a0, a1, b0, b1;
    {
      const int r0 = wm * 32 + fm;
      v8bf lo = *(const v8bf*)&As[r0][akb];
      v8bf hi = *(const v8bf*)&As[r0][akb + 8];
#pragma unroll
      for (int i = 0; i < 8; ++i) { a0[i] = lo[i]; a0[8 + i] = hi[i]; }
    }
    {
      const int r1 = wm * 32 + 16 + fm;
      v8bf lo = *(const v8bf*)&As[r1][akb];
      v8bf hi = *(const v8bf*)&As[r1][akb + 8];
#pragma unroll
      for (int i = 0; i < 8; ++i) { a1[i] = lo[i]; a1[8 + i] = hi[i]; }
    }
    {
      const int c0 = wn * 32 + fm;
      v8bf lo = *(const v8bf*)&Bs[c0][bkb];
      v8bf hi = *(const v8bf*)&Bs[c0][bkb + 8];
#pragma unroll
      for (int i = 0; i < 8; ++i) { b0[i] = lo[i]; b0[8 + i] = hi[i]; }
    }
    {
      const int c1 = wn * 32 + 16 + fm;
      v8bf lo = *(const v8bf*)&Bs[c1][bkb];
      v8bf hi = *(const v8bf*)&Bs[c1][bkb + 8];
#pragma unroll
      for (int i = 0; i < 8; ++i) { b1[i] = lo[i]; b1[8 + i] = hi[i]; }
    }
    acc00 = __builtin_amdgcn_wmma_f32_16x16x32_bf16(false, a0, false, b0, (short)0, acc00, false, false);
    acc01 = __builtin_amdgcn_wmma_f32_16x16x32_bf16(false, a0, false, b1, (short)0, acc01, false, false);
    acc10 = __builtin_amdgcn_wmma_f32_16x16x32_bf16(false, a1, false, b0, (short)0, acc10, false, false);
    acc11 = __builtin_amdgcn_wmma_f32_16x16x32_bf16(false, a1, false, b1, (short)0, acc11, false, false);
    __syncthreads();
  }
  // ---- epilogue ----
#pragma unroll
  for (int mi = 0; mi < 2; ++mi) {
#pragma unroll
    for (int ni = 0; ni < 2; ++ni) {
      v8f acc = (mi == 0) ? (ni == 0 ? acc00 : acc01) : (ni == 0 ? acc10 : acc11);
      const int col = bn + wn * 32 + ni * 16 + fm;
      if (col >= p.N) continue;
      float bval = 0.0f;
      if (BIAS) bval = p.bias[col];
#pragma unroll
      for (int v = 0; v < 8; ++v) {
        const int row = bm + wm * 32 + mi * 16 + ((lane < 16) ? v : v + 8);
        if (row >= p.M) continue;
        float xv = acc[v] * p.alpha;
        if (BIAS) xv += bval;
        if (GELU) xv = 0.5f * xv * (1.0f + erff(xv * 0.70710678118654752f));
        const long ci = cOff + (long)row * p.ldc + col;
        if (RESID) xv += p.resid[ci];
        if (SF32)  p.Cf[ci] = xv;
        if (SBF16) p.Cb[ci] = (__bf16)xv;
      }
    }
  }
}

// ---------------- elementwise / reduction kernels ----------------
__global__ void convert_bf16_k(const float* in, __bf16* out, long n) {
  long i = (long)blockIdx.x * 256 + threadIdx.x;
  if (i < n) out[i] = (__bf16)in[i];
}

// conv_w [HID][768] -> out[k][d] (B matrix for patch GEMM)
__global__ void transpose_convert_k(const float* in, __bf16* out, int rows, int cols) {
  long i = (long)blockIdx.x * 256 + threadIdx.x;
  if (i >= (long)rows * cols) return;
  int r = (int)(i / cols), c = (int)(i % cols);
  out[(long)c * rows + r] = (__bf16)in[i];
}

// wq[l] [NH][HID][DH] -> out[d][h*64+k]  (bias vector [NH*DH] matches columns)
__global__ void pack_qkv_k(const float* in, __bf16* out) {
  long i = (long)blockIdx.x * 256 + threadIdx.x;
  if (i >= (long)HIDD * HIDD) return;
  int d = (int)(i / HIDD), o = (int)(i % HIDD);
  int hh = o >> 6, kk = o & 63;
  out[i] = (__bf16)in[((long)hh * HIDD + d) * DHD + kk];
}

// im2col: X[t][c*256+p*16+q] = x[b,c,i*16+p,j*16+q]
__global__ void patchify_k(const float* x, __bf16* out) {
  long i = (long)blockIdx.x * 256 + threadIdx.x;
  if (i >= (long)NPATCH * HIDD) return;
  int f = (int)(i % HIDD);
  int t = (int)(i / HIDD);
  int b = t / NPP, ij = t % NPP, pi = ij / 14, pj = ij % 14;
  int c = f >> 8, rr = (f >> 4) & 15, q = f & 15;
  out[i] = (__bf16)x[(((long)(b * 3 + c) * 224) + (pi * 16 + rr)) * 224 + (pj * 16 + q)];
}

// h[b,0,:] = cls + pos[0]; h[b,1+ij,:] = Hpatch + pos[1+ij]
__global__ void assemble_k(const float* Hp, const float* cls, const float* pos, float* h) {
  long i = (long)blockIdx.x * 256 + threadIdx.x;
  if (i >= (long)TOK * HIDD) return;
  int d = (int)(i % HIDD);
  long t = i / HIDD;
  int b = (int)(t / SEQ), s = (int)(t % SEQ);
  float v;
  if (s == 0) v = cls[d] + pos[d];
  else        v = Hp[((long)b * NPP + (s - 1)) * HIDD + d] + pos[(long)s * HIDD + d];
  h[i] = v;
}

__global__ __launch_bounds__(256) void layernorm_k(const float* h, const float* g,
                                                   const float* b, __bf16* out) {
  const int t = blockIdx.x;
  const float* x = h + (long)t * HIDD;
  __bf16* o = out + (long)t * HIDD;
  __shared__ float red[256];
  const int tid = threadIdx.x;
  float s = 0.f;
  for (int i = tid; i < HIDD; i += 256) s += x[i];
  red[tid] = s; __syncthreads();
  for (int st = 128; st > 0; st >>= 1) { if (tid < st) red[tid] += red[tid + st]; __syncthreads(); }
  const float mu = red[0] * (1.0f / HIDD); __syncthreads();
  float v = 0.f;
  for (int i = tid; i < HIDD; i += 256) { float d = x[i] - mu; v += d * d; }
  red[tid] = v; __syncthreads();
  for (int st = 128; st > 0; st >>= 1) { if (tid < st) red[tid] += red[tid + st]; __syncthreads(); }
  const float inv = rsqrtf(red[0] * (1.0f / HIDD) + 1e-5f);
  for (int i = tid; i < HIDD; i += 256) o[i] = (__bf16)((x[i] - mu) * inv * g[i] + b[i]);
}

// one block per attention row; rows are SEQP-strided; pad cols [n, SEQP) zeroed
__global__ __launch_bounds__(256) void softmax_k(const float* sc, __bf16* pr, int n) {
  const long r = blockIdx.x;
  const float* x = sc + r * SEQP;
  __bf16* o = pr + r * SEQP;
  __shared__ float red[256];
  const int tid = threadIdx.x;
  float m = -1e30f;
  for (int i = tid; i < n; i += 256) m = fmaxf(m, x[i]);
  red[tid] = m; __syncthreads();
  for (int st = 128; st > 0; st >>= 1) { if (tid < st) red[tid] = fmaxf(red[tid], red[tid + st]); __syncthreads(); }
  m = red[0]; __syncthreads();
  float s = 0.f;
  for (int i = tid; i < n; i += 256) s += expf(x[i] - m);
  red[tid] = s; __syncthreads();
  for (int st = 128; st > 0; st >>= 1) { if (tid < st) red[tid] += red[tid + st]; __syncthreads(); }
  const float inv = 1.0f / red[0];
  for (int i = tid; i < SEQP; i += 256)
    o[i] = (i < n) ? (__bf16)(expf(x[i] - m) * inv) : (__bf16)0.0f;
}

// kT[b][h][kk][s] = K[b][s][h][kk], s-dim padded to SEQP
__global__ void kT_k(const __bf16* kin, __bf16* kT) {
  long i = (long)blockIdx.x * 256 + threadIdx.x;
  const long n = (long)BATCHN * NHD * DHD * SEQ;
  if (i >= n) return;
  int s  = (int)(i % SEQ);
  long r = i / SEQ;
  int kk = (int)(r % DHD); r /= DHD;
  int hh = (int)(r % NHD);
  int b  = (int)(r / NHD);
  kT[(((long)(b * NHD + hh)) * DHD + kk) * SEQP + s] =
      kin[((long)(b * SEQ + s)) * HIDD + hh * DHD + kk];
}

__global__ void cls_k(const float* h, __bf16* out) {
  int i = blockIdx.x * 256 + threadIdx.x;
  if (i >= BATCHN * HIDD) return;
  int b = i / HIDD, d = i % HIDD;
  out[i] = (__bf16)h[(long)b * SEQ * HIDD + d];
}

// ---------------- host-side helpers ----------------
static inline int cdiv(int a, int b) { return (a + b - 1) / b; }
static inline dim3 ew_grid(long n) { return dim3((unsigned)((n + 255) / 256)); }

enum { F_BIAS = 1, F_GELU = 2, F_RESID = 4, F_F32 = 8, F_BF16 = 16 };

static void launch_gemm(hipStream_t st, int flags,
                        const __bf16* A, const __bf16* B, const float* bias,
                        const float* resid, float* Cf, __bf16* Cb,
                        int M, int N, int K, int lda, int ldb, int ldc,
                        int batch, int bdiv,
                        long aS1, long aS2, long bS1, long bS2, long cS1, long cS2,
                        float alpha) {
  GemmP p{A, B, bias, resid, Cf, Cb, M, N, K, lda, ldb, ldc,
          bdiv, aS1, aS2, bS1, bS2, cS1, cS2, alpha};
  dim3 g(cdiv(N, 64), cdiv(M, 64), batch), blk(128);
  if      (flags == (F_BIAS | F_BF16))           gemm_wmma_k<true,  false, false, false, true ><<<g, blk, 0, st>>>(p);
  else if (flags == F_F32)                       gemm_wmma_k<false, false, false, true,  false><<<g, blk, 0, st>>>(p);
  else if (flags == F_BF16)                      gemm_wmma_k<false, false, false, false, true ><<<g, blk, 0, st>>>(p);
  else if (flags == (F_BIAS | F_RESID | F_F32))  gemm_wmma_k<true,  false, true,  true,  false><<<g, blk, 0, st>>>(p);
  else if (flags == (F_BIAS | F_F32))            gemm_wmma_k<true,  false, false, true,  false><<<g, blk, 0, st>>>(p);
  else if (flags == (F_BIAS | F_GELU | F_BF16))  gemm_wmma_k<true,  true,  false, false, true ><<<g, blk, 0, st>>>(p);
}

extern "C" void kernel_launch(void* const* d_in, const int* in_sizes, int n_in,
                              void* d_out, int out_size, void* d_ws, size_t ws_size,
                              hipStream_t stream) {
  (void)in_sizes; (void)n_in; (void)out_size; (void)ws_size;
  const float* x       = (const float*)d_in[0];
  const float* conv_w  = (const float*)d_in[1];
  const float* conv_b  = (const float*)d_in[2];
  const float* cls_t   = (const float*)d_in[3];
  const float* pos_emb = (const float*)d_in[4];
  const float* ln1_s   = (const float*)d_in[5];
  const float* ln1_b   = (const float*)d_in[6];
  const float* wq      = (const float*)d_in[7];
  const float* bq      = (const float*)d_in[8];
  const float* wk      = (const float*)d_in[9];
  const float* bk      = (const float*)d_in[10];
  const float* wv      = (const float*)d_in[11];
  const float* bv      = (const float*)d_in[12];
  const float* wo      = (const float*)d_in[13];
  const float* bo      = (const float*)d_in[14];
  const float* ln2_s   = (const float*)d_in[15];
  const float* ln2_b   = (const float*)d_in[16];
  const float* w1      = (const float*)d_in[17];
  const float* b1      = (const float*)d_in[18];
  const float* w2      = (const float*)d_in[19];
  const float* b2      = (const float*)d_in[20];
  const float* head_w  = (const float*)d_in[21];
  const float* head_b  = (const float*)d_in[22];

  // workspace bump allocator (256B aligned)
  char* base = (char*)d_ws;
  size_t off = 0;
  auto alloc = [&](size_t bytes) -> void* {
    off = (off + 255) & ~(size_t)255;
    void* p = base + off; off += bytes; return p;
  };
  __bf16* Xp     = (__bf16*)alloc((size_t)NPATCH * HIDD * 2);
  float*  Hp     = (float*) alloc((size_t)NPATCH * HIDD * 4);
  float*  h      = (float*) alloc((size_t)TOK * HIDD * 4);
  __bf16* hn     = (__bf16*)alloc((size_t)TOK * HIDD * 2);
  __bf16* qb     = (__bf16*)alloc((size_t)TOK * HIDD * 2);
  __bf16* kbuf   = (__bf16*)alloc((size_t)TOK * HIDD * 2);
  __bf16* vbuf   = (__bf16*)alloc((size_t)TOK * HIDD * 2);
  __bf16* kT     = (__bf16*)alloc((size_t)BATCHN * NHD * DHD * SEQP * 2);
  float*  scores = (float*) alloc((size_t)BATCHN * NHD * SEQ * SEQP * 4);
  __bf16* probs  = (__bf16*)alloc((size_t)BATCHN * NHD * SEQ * SEQP * 2);
  __bf16* att    = (__bf16*)alloc((size_t)TOK * HIDD * 2);
  __bf16* mbuf   = (__bf16*)alloc((size_t)TOK * INTERD * 2);
  __bf16* cls32  = (__bf16*)alloc((size_t)BATCHN * HIDD * 2);
  __bf16* convWT = (__bf16*)alloc((size_t)HIDD * HIDD * 2);
  __bf16* wqP    = (__bf16*)alloc((size_t)HIDD * HIDD * 2);
  __bf16* wkP    = (__bf16*)alloc((size_t)HIDD * HIDD * 2);
  __bf16* wvP    = (__bf16*)alloc((size_t)HIDD * HIDD * 2);
  __bf16* woC    = (__bf16*)alloc((size_t)HIDD * HIDD * 2);
  __bf16* w1C    = (__bf16*)alloc((size_t)HIDD * INTERD * 2);
  __bf16* w2C    = (__bf16*)alloc((size_t)INTERD * HIDD * 2);
  __bf16* headC  = (__bf16*)alloc((size_t)HIDD * NCLS * 2);

  // ---- patch embedding ----
  transpose_convert_k<<<ew_grid((long)HIDD * HIDD), 256, 0, stream>>>(conv_w, convWT, HIDD, HIDD);
  convert_bf16_k<<<ew_grid((long)HIDD * NCLS), 256, 0, stream>>>(head_w, headC, (long)HIDD * NCLS);
  patchify_k<<<ew_grid((long)NPATCH * HIDD), 256, 0, stream>>>(x, Xp);
  launch_gemm(stream, F_BIAS | F_F32, Xp, convWT, conv_b, nullptr, Hp, nullptr,
              NPATCH, HIDD, HIDD, HIDD, HIDD, HIDD, 1, 1, 0, 0, 0, 0, 0, 0, 1.0f);
  assemble_k<<<ew_grid((long)TOK * HIDD), 256, 0, stream>>>(Hp, cls_t, pos_emb, h);

  // ---- transformer layers ----
  for (int l = 0; l < LYR; ++l) {
    const long wl = (long)l * NHD * HIDD * DHD;
    pack_qkv_k<<<ew_grid((long)HIDD * HIDD), 256, 0, stream>>>(wq + wl, wqP);
    pack_qkv_k<<<ew_grid((long)HIDD * HIDD), 256, 0, stream>>>(wk + wl, wkP);
    pack_qkv_k<<<ew_grid((long)HIDD * HIDD), 256, 0, stream>>>(wv + wl, wvP);
    convert_bf16_k<<<ew_grid((long)HIDD * HIDD), 256, 0, stream>>>(wo + (long)l * HIDD * HIDD, woC, (long)HIDD * HIDD);
    convert_bf16_k<<<ew_grid((long)HIDD * INTERD), 256, 0, stream>>>(w1 + (long)l * HIDD * INTERD, w1C, (long)HIDD * INTERD);
    convert_bf16_k<<<ew_grid((long)INTERD * HIDD), 256, 0, stream>>>(w2 + (long)l * INTERD * HIDD, w2C, (long)INTERD * HIDD);

    // LN1 -> bf16
    layernorm_k<<<TOK, 256, 0, stream>>>(h, ln1_s + (long)l * HIDD, ln1_b + (long)l * HIDD, hn);

    // Q, K, V projections (fused heads: [TOK,768] x [768,768])
    launch_gemm(stream, F_BIAS | F_BF16, hn, wqP, bq + (long)l * HIDD, nullptr, nullptr, qb,
                TOK, HIDD, HIDD, HIDD, HIDD, HIDD, 1, 1, 0, 0, 0, 0, 0, 0, 1.0f);
    launch_gemm(stream, F_BIAS | F_BF16, hn, wkP, bk + (long)l * HIDD, nullptr, nullptr, kbuf,
                TOK, HIDD, HIDD, HIDD, HIDD, HIDD, 1, 1, 0, 0, 0, 0, 0, 0, 1.0f);
    launch_gemm(stream, F_BIAS | F_BF16, hn, wvP, bv + (long)l * HIDD, nullptr, nullptr, vbuf,
                TOK, HIDD, HIDD, HIDD, HIDD, HIDD, 1, 1, 0, 0, 0, 0, 0, 0, 1.0f);

    kT_k<<<ew_grid((long)BATCHN * NHD * DHD * SEQ), 256, 0, stream>>>(kbuf, kT);

    // scores = (Q @ K^T) / 8 : batched over (b,h); N padded to SEQP (pad cols unused)
    launch_gemm(stream, F_F32, qb, kT, nullptr, nullptr, scores, nullptr,
                SEQ, SEQP, DHD, HIDD, SEQP, SEQP,
                BATCHN * NHD, NHD,
                (long)SEQ * HIDD, DHD,
                (long)NHD * DHD * SEQP, (long)DHD * SEQP,
                (long)NHD * SEQ * SEQP, (long)SEQ * SEQP,
                0.125f);

    softmax_k<<<BATCHN * NHD * SEQ, 256, 0, stream>>>(scores, probs, SEQ);

    // att = P @ V; K padded to SEQP (pad coefficients are exactly zero)
    launch_gemm(stream, F_BF16, probs, vbuf, nullptr, nullptr, nullptr, att,
                SEQ, DHD, SEQP, SEQP, HIDD, HIDD,
                BATCHN * NHD, NHD,
                (long)NHD * SEQ * SEQP, (long)SEQ * SEQP,
                (long)SEQ * HIDD, DHD,
                (long)SEQ * HIDD, DHD,
                1.0f);

    // h += att @ wo + bo
    launch_gemm(stream, F_BIAS | F_RESID | F_F32, att, woC, bo + (long)l * HIDD, h, h, nullptr,
                TOK, HIDD, HIDD, HIDD, HIDD, HIDD, 1, 1, 0, 0, 0, 0, 0, 0, 1.0f);

    // MLP
    layernorm_k<<<TOK, 256, 0, stream>>>(h, ln2_s + (long)l * HIDD, ln2_b + (long)l * HIDD, hn);
    launch_gemm(stream, F_BIAS | F_GELU | F_BF16, hn, w1C, b1 + (long)l * INTERD, nullptr, nullptr, mbuf,
                TOK, INTERD, HIDD, HIDD, INTERD, INTERD, 1, 1, 0, 0, 0, 0, 0, 0, 1.0f);
    launch_gemm(stream, F_BIAS | F_RESID | F_F32, mbuf, w2C, b2 + (long)l * HIDD, h, h, nullptr,
                TOK, HIDD, INTERD, INTERD, HIDD, HIDD, 1, 1, 0, 0, 0, 0, 0, 0, 1.0f);
  }

  // ---- classifier head on [CLS] ----
  cls_k<<<ew_grid((long)BATCHN * HIDD), 256, 0, stream>>>(h, cls32);
  launch_gemm(stream, F_BIAS | F_F32, cls32, headC, head_b, nullptr, (float*)d_out, nullptr,
              BATCHN, NCLS, HIDD, HIDD, NCLS, NCLS, 1, 1, 0, 0, 0, 0, 0, 0, 1.0f);
}
